// MultiHeadedAttention_26749056319718
// MI455X (gfx1250) — compile-verified, matrix operand order checked by reading, not run on hardware
//
#include <hip/hip_runtime.h>

// ---------------------------------------------------------------------------
// MultiHeadedAttention with learned memory slots — MI455X (gfx1250) wave32
// bf16 WMMA (v_wmma_f32_16x16x32_bf16) for all GEMMs, flash-attention softmax,
// block-cooperative async LDS staging of K/V tiles (ASYNCcnt path).
// ---------------------------------------------------------------------------

typedef __attribute__((ext_vector_type(16))) __bf16 v16bf;
typedef __attribute__((ext_vector_type(8)))  __bf16 v8bf;
typedef __attribute__((ext_vector_type(8)))  float  v8f;
typedef __attribute__((ext_vector_type(4)))  int    v4i;

#define B_   16
#define NQ_  1024
#define NK_  1024
#define DM_  512
#define H_   8
#define DK_  64
#define M_   40
#define NKP_ 1088   /* nk + M padded to 34*32 */
#define NKM_ 1064   /* nk + M (valid columns) */

// ---- workspace layout (bytes) ---------------------------------------------
#define OFF_XQ   ((size_t)0)                       // 16*1024*512 bf16
#define OFF_XK   (OFF_XQ + (size_t)B_*NQ_*DM_*2)
#define OFF_XV   (OFF_XK + (size_t)B_*NQ_*DM_*2)
#define OFF_WQT  (OFF_XV + (size_t)B_*NQ_*DM_*2)   // 512*512 bf16 each
#define OFF_WKT  (OFF_WQT + (size_t)DM_*DM_*2)
#define OFF_WVT  (OFF_WKT + (size_t)DM_*DM_*2)
#define OFF_WOT  (OFF_WVT + (size_t)DM_*DM_*2)
#define OFF_QH   (OFF_WOT + (size_t)DM_*DM_*2)     // [b,h,1024,64] bf16
#define OFF_KH   (OFF_QH  + (size_t)B_*H_*NQ_*DK_*2)   // [b,h,1088,64] bf16
#define OFF_VT   (OFF_KH  + (size_t)B_*H_*NKP_*DK_*2)  // [b,h,64,1088] bf16
#define OFF_CTX  (OFF_VT  + (size_t)B_*H_*DK_*NKP_*2)  // [b,1024,512] bf16
#define OFF_Y    ((size_t)0)                       // f32 [b,1024,512] aliases Xq+Xk

// ---- async load-to-LDS (gfx1250), gated on builtin availability ------------
#if __has_builtin(__builtin_amdgcn_global_load_async_to_lds_b128) && \
    __has_builtin(__builtin_amdgcn_s_wait_asynccnt)
#define USE_ASYNC 1
#else
#define USE_ASYNC 0
#endif

typedef __attribute__((address_space(1))) v4i gv4i;   // global int4
typedef __attribute__((address_space(3))) v4i lv4i;   // LDS int4

// ---- helpers ---------------------------------------------------------------
__device__ __forceinline__ unsigned short f2bf(float f) {
  unsigned u = __builtin_bit_cast(unsigned, f);
  u = u + 0x7FFFu + ((u >> 16) & 1u);              // round-to-nearest-even
  return (unsigned short)(u >> 16);
}

__device__ __forceinline__ v16bf mkfrag(const unsigned short* lo,
                                        const unsigned short* hi) {
  v8bf a = *reinterpret_cast<const v8bf*>(lo);
  v8bf b = *reinterpret_cast<const v8bf*>(hi);
  return __builtin_shufflevector(a, b, 0,1,2,3,4,5,6,7,8,9,10,11,12,13,14,15);
}

// fragment with K-halves at base[k0 + half*8] / base[k0 + 16 + half*8]
__device__ __forceinline__ v16bf ldfrag(const unsigned short* base, int k0, int half) {
  return mkfrag(base + k0 + half * 8, base + k0 + 16 + half * 8);
}

__device__ __forceinline__ v8f wmma_bf16(v16bf a, v16bf b, v8f c) {
  return __builtin_amdgcn_wmma_f32_16x16x32_bf16(false, a, false, b,
                                                 (short)0, c, false, false);
}

// 16-byte global -> LDS copy (async when available, sync fallback)
__device__ __forceinline__ void cp16(const unsigned short* g, unsigned short* l) {
#if USE_ASYNC
  __builtin_amdgcn_global_load_async_to_lds_b128((gv4i*)(void*)g, (lv4i*)l, 0, 0);
#else
  *reinterpret_cast<v8bf*>(l) = *reinterpret_cast<const v8bf*>(g);
#endif
}
__device__ __forceinline__ void cp16_wait_keep2() {
#if USE_ASYNC
  __builtin_amdgcn_s_wait_asynccnt(2);
#endif
}
__device__ __forceinline__ void cp16_wait_all() {
#if USE_ASYNC
  __builtin_amdgcn_s_wait_asynccnt(0);
#endif
}

// ---- kernel 1a: f32 -> bf16, 8 elements / thread ---------------------------
__global__ void conv_bf16_k(const float* __restrict__ src,
                            unsigned short* __restrict__ dst) {
  int i = blockIdx.x * blockDim.x + threadIdx.x;
  const float4* s = reinterpret_cast<const float4*>(src);
  float4 a = s[2 * i], b = s[2 * i + 1];
  uint4 o;
  o.x = (unsigned)f2bf(a.x) | ((unsigned)f2bf(a.y) << 16);
  o.y = (unsigned)f2bf(a.z) | ((unsigned)f2bf(a.w) << 16);
  o.z = (unsigned)f2bf(b.x) | ((unsigned)f2bf(b.y) << 16);
  o.w = (unsigned)f2bf(b.z) | ((unsigned)f2bf(b.w) << 16);
  reinterpret_cast<uint4*>(dst)[i] = o;
}

// ---- kernel 1b: transpose 512x512 weight, f32 -> bf16 ----------------------
__global__ void transpose_w_k(const float* __restrict__ W,
                              unsigned short* __restrict__ Wt) {
  int idx = blockIdx.x * blockDim.x + threadIdx.x;   // 512*512
  int k = idx >> 9, n = idx & 511;
  Wt[(size_t)n * DM_ + k] = f2bf(W[idx]);
}

// ---- kernel 1c: fill memory rows of Kh and memory cols of Vt (+zero pad) ---
__global__ void fill_mem_k(const float* __restrict__ memK,
                           const float* __restrict__ memV,
                           unsigned short* __restrict__ Kh,
                           unsigned short* __restrict__ Vt) {
  int idx = blockIdx.x * blockDim.x + threadIdx.x;   // b*h*64*64
  int d  = idx & 63;
  int m  = (idx >> 6) & 63;                          // row 1024+m
  int bh = idx >> 12;
  int h  = bh & 7;
  float kv = 0.f, vv = 0.f;
  if (m < M_) {
    kv = 8.0f       * memK[(size_t)m * (H_*DK_) + h * DK_ + d];   // sqrt(DK)=8
    vv = 6.3245553f * memV[(size_t)m * (H_*DK_) + h * DK_ + d];   // sqrt(M)
  }
  Kh[((size_t)bh * NKP_ + NK_ + m) * DK_ + d] = f2bf(kv);
  Vt[((size_t)bh * DK_ + d) * NKP_ + NK_ + m] = f2bf(vv);
}

// ---- kernel 2: projection GEMM  out = X(bf16) @ Wt^T + bias ----------------
// grid (8, 128), block 256 (8 waves). Wave = 16 rows x 64 cols.
// Software-pipelined (unroll 2 => register double-buffer without rotation moves).
__global__ void __launch_bounds__(256)
gemm_proj_k(const unsigned short* __restrict__ X,
            const unsigned short* __restrict__ Wt,
            const float* __restrict__ bias,
            unsigned short* __restrict__ dst,
            long hS, long qS, long dS) {
  const int lane = threadIdx.x & 31;
  const int w = threadIdx.x >> 5;
  const int nn = lane & 15, half = lane >> 4;
  const int rowBase = blockIdx.y * 128 + w * 16;
  const int colBase = blockIdx.x * 64;

  v8f acc0 = {}, acc1 = {}, acc2 = {}, acc3 = {};
  const unsigned short* xrow = X + (size_t)(rowBase + nn) * DM_;
  const unsigned short* w0 = Wt + (size_t)(colBase +  0 + nn) * DM_;
  const unsigned short* w1 = Wt + (size_t)(colBase + 16 + nn) * DM_;
  const unsigned short* w2 = Wt + (size_t)(colBase + 32 + nn) * DM_;
  const unsigned short* w3 = Wt + (size_t)(colBase + 48 + nn) * DM_;

  v16bf a  = ldfrag(xrow, 0, half);
  v16bf f0 = ldfrag(w0, 0, half), f1 = ldfrag(w1, 0, half);
  v16bf f2 = ldfrag(w2, 0, half), f3 = ldfrag(w3, 0, half);

  #pragma unroll 2
  for (int k0 = 32; k0 < DM_; k0 += 32) {
    v16bf an = ldfrag(xrow, k0, half);
    v16bf g0 = ldfrag(w0, k0, half), g1 = ldfrag(w1, k0, half);
    v16bf g2 = ldfrag(w2, k0, half), g3 = ldfrag(w3, k0, half);
    acc0 = wmma_bf16(a, f0, acc0);
    acc1 = wmma_bf16(a, f1, acc1);
    acc2 = wmma_bf16(a, f2, acc2);
    acc3 = wmma_bf16(a, f3, acc3);
    a = an; f0 = g0; f1 = g1; f2 = g2; f3 = g3;
  }
  acc0 = wmma_bf16(a, f0, acc0);
  acc1 = wmma_bf16(a, f1, acc1);
  acc2 = wmma_bf16(a, f2, acc2);
  acc3 = wmma_bf16(a, f3, acc3);

  #pragma unroll
  for (int t = 0; t < 4; t++) {
    v8f acc = (t == 0) ? acc0 : (t == 1) ? acc1 : (t == 2) ? acc2 : acc3;
    #pragma unroll
    for (int i = 0; i < 8; i++) {
      int row = rowBase + i + 8 * half;
      int col = colBase + t * 16 + nn;
      float v = acc[i] + bias[col];
      int b = row >> 10, q = row & 1023;
      int h = col >> 6,  d = col & 63;
      dst[(size_t)(b * H_ + h) * hS + (size_t)q * qS + (size_t)d * dS] = f2bf(v);
    }
  }
}

// ---- kernel 3: flash attention ---------------------------------------------
// grid 1024 = b(16)*h(8)*qtile(8); block 256 = 8 waves; wave = 16 q rows.
// K(32x64) and V^T(64x32) tiles staged cooperatively into LDS via async
// load-to-LDS, double-buffered across key-block iterations.
__global__ void __launch_bounds__(256)
attn_k(const unsigned short* __restrict__ Qh,
       const unsigned short* __restrict__ Kh,
       const unsigned short* __restrict__ Vt,
       const float* __restrict__ attW,
       unsigned short* __restrict__ ctx) {
  __shared__ __align__(16) unsigned short kt[2][32][64];    // 8 KB
  __shared__ __align__(16) unsigned short vt[2][64][32];    // 8 KB
  __shared__ __align__(16) unsigned short plds[8][16][40];  // 10 KB

  const int tid = threadIdx.x;
  const int lane = tid & 31, w = tid >> 5;
  const int nn = lane & 15, half = lane >> 4;
  const int blk = blockIdx.x;
  const int b = blk >> 6, h = (blk >> 3) & 7, qt = blk & 7;
  const int bh = b * H_ + h;
  const int qBase = qt * 128 + w * 16;

  const unsigned short* Khb = Kh + (size_t)bh * NKP_ * DK_;
  const unsigned short* Vtb = Vt + (size_t)bh * DK_ * NKP_;

  // per-thread 16B chunks of the K / V^T tiles
  const int kr = tid >> 3, kc = (tid & 7) * 8;   // ktile: 32 x 64
  const int vr = tid >> 2, vc = (tid & 3) * 8;   // vtile: 64 x 32

  // Q fragments (16x64 -> two 16x32 A-frags)
  const unsigned short* qp = Qh + ((size_t)bh * NQ_ + qBase + nn) * DK_;
  v16bf qa0 = ldfrag(qp, 0, half);
  v16bf qa1 = ldfrag(qp, 32, half);

  v8f o0 = {}, o1 = {}, o2 = {}, o3 = {};
  float m_[8], l_[8];
  #pragma unroll
  for (int i = 0; i < 8; i++) { m_[i] = -3.0e38f; l_[i] = 0.f; }

  const float* wbase = attW + ((size_t)b << 20);

  // prologue: stage tile 0
  cp16(Khb + (size_t)kr * DK_ + kc, &kt[0][kr][kc]);
  cp16(Vtb + (size_t)vr * NKP_ + vc, &vt[0][vr][vc]);

  #pragma unroll 1
  for (int it = 0; it < NKP_ / 32; ++it) {
    const int par = it & 1;
    const int kb = it * 32;
    if (it + 1 < NKP_ / 32) {          // prefetch next tile into other buffer
      const int kn = kb + 32;
      cp16(Khb + (size_t)(kn + kr) * DK_ + kc, &kt[par ^ 1][kr][kc]);
      cp16(Vtb + (size_t)vr * NKP_ + kn + vc, &vt[par ^ 1][vr][vc]);
      cp16_wait_keep2();               // batch `it` landed; next stays in flight
    } else {
      cp16_wait_all();
    }
    __syncthreads();

    const unsigned short (*ktp)[64] = kt[par];
    const unsigned short (*vtp)[32] = vt[par];

    // --- S = (Q K^T) for 32 keys: two 16x16 C tiles, K-frags from LDS ---
    v8f s0 = {}, s1 = {};
    s0 = wmma_bf16(qa0, ldfrag(&ktp[nn][0],       0, half), s0);
    s0 = wmma_bf16(qa1, ldfrag(&ktp[nn][0],      32, half), s0);
    s1 = wmma_bf16(qa0, ldfrag(&ktp[16 + nn][0],  0, half), s1);
    s1 = wmma_bf16(qa1, ldfrag(&ktp[16 + nn][0], 32, half), s1);

    // --- scale 1/sqrt(dk), modulate with attention_weights, mask padding ---
    const int k0i = kb + nn, k1i = kb + 16 + nn;
    #pragma unroll
    for (int i = 0; i < 8; i++) {
      int q = qBase + i + 8 * half;
      float v0 = s0[i] * 0.125f;
      if (k0i < NK_)        v0 *= wbase[((size_t)q << 10) + k0i];
      else if (k0i >= NKM_) v0 = -3.0e38f;
      s0[i] = v0;
      float v1 = s1[i] * 0.125f;
      if (k1i < NK_)        v1 *= wbase[((size_t)q << 10) + k1i];
      else if (k1i >= NKM_) v1 = -3.0e38f;
      s1[i] = v1;
    }
    // --- online softmax (row stats replicated across each 16-lane half) ---
    float alpha[8];
    #pragma unroll
    for (int i = 0; i < 8; i++) {
      float t = fmaxf(s0[i], s1[i]);
      t = fmaxf(t, __shfl_xor(t, 1));
      t = fmaxf(t, __shfl_xor(t, 2));
      t = fmaxf(t, __shfl_xor(t, 4));
      t = fmaxf(t, __shfl_xor(t, 8));
      float nm = fmaxf(m_[i], t);
      float p0 = __expf(s0[i] - nm), p1 = __expf(s1[i] - nm);
      s0[i] = p0; s1[i] = p1;
      float rs = p0 + p1;
      rs += __shfl_xor(rs, 1);
      rs += __shfl_xor(rs, 2);
      rs += __shfl_xor(rs, 4);
      rs += __shfl_xor(rs, 8);
      float a = __expf(m_[i] - nm);
      l_[i] = l_[i] * a + rs;
      m_[i] = nm;
      alpha[i] = a;
    }
    #pragma unroll
    for (int i = 0; i < 8; i++) {
      o0[i] *= alpha[i]; o1[i] *= alpha[i];
      o2[i] *= alpha[i]; o3[i] *= alpha[i];
    }
    // --- P (C layout) -> LDS -> A-fragment layout (same-wave DScnt dep) ---
    #pragma unroll
    for (int i = 0; i < 8; i++) {
      plds[w][i + 8 * half][nn]      = f2bf(s0[i]);
      plds[w][i + 8 * half][16 + nn] = f2bf(s1[i]);
    }
    v16bf pa = mkfrag(&plds[w][nn][half * 8], &plds[w][nn][16 + half * 8]);
    // --- O += P @ V  (V^T fragments from LDS: contiguous over keys) ---
    o0 = wmma_bf16(pa, ldfrag(&vtp[ 0 + nn][0], 0, half), o0);
    o1 = wmma_bf16(pa, ldfrag(&vtp[16 + nn][0], 0, half), o1);
    o2 = wmma_bf16(pa, ldfrag(&vtp[32 + nn][0], 0, half), o2);
    o3 = wmma_bf16(pa, ldfrag(&vtp[48 + nn][0], 0, half), o3);

    __syncthreads();   // protect buf[par] from being restaged next iteration
  }

  // --- normalize by softmax denominator; write ctx [b, q, h*64 + dv] -------
  unsigned short* cbase = ctx + (size_t)b * NQ_ * DM_ + h * DK_;
  #pragma unroll
  for (int i = 0; i < 8; i++) {
    float inv = 1.0f / l_[i];
    int q = qBase + i + 8 * half;
    unsigned short* cp = cbase + (size_t)q * DM_;
    cp[ 0 + nn] = f2bf(o0[i] * inv);
    cp[16 + nn] = f2bf(o1[i] * inv);
    cp[32 + nn] = f2bf(o2[i] * inv);
    cp[48 + nn] = f2bf(o3[i] * inv);
  }
}

// ---- kernel 4: output projection + bias + residual -> f32 ------------------
__global__ void __launch_bounds__(256)
gemm_out_k(const unsigned short* __restrict__ X,
           const unsigned short* __restrict__ Wt,
           const float* __restrict__ bias,
           const float* __restrict__ resid,
           float* __restrict__ y) {
  const int lane = threadIdx.x & 31;
  const int w = threadIdx.x >> 5;
  const int nn = lane & 15, half = lane >> 4;
  const int rowBase = blockIdx.y * 128 + w * 16;
  const int colBase = blockIdx.x * 64;

  v8f acc0 = {}, acc1 = {}, acc2 = {}, acc3 = {};
  const unsigned short* xrow = X + (size_t)(rowBase + nn) * DM_;
  const unsigned short* w0 = Wt + (size_t)(colBase +  0 + nn) * DM_;
  const unsigned short* w1 = Wt + (size_t)(colBase + 16 + nn) * DM_;
  const unsigned short* w2 = Wt + (size_t)(colBase + 32 + nn) * DM_;
  const unsigned short* w3 = Wt + (size_t)(colBase + 48 + nn) * DM_;

  v16bf a  = ldfrag(xrow, 0, half);
  v16bf f0 = ldfrag(w0, 0, half), f1 = ldfrag(w1, 0, half);
  v16bf f2 = ldfrag(w2, 0, half), f3 = ldfrag(w3, 0, half);

  #pragma unroll 2
  for (int k0 = 32; k0 < DM_; k0 += 32) {
    v16bf an = ldfrag(xrow, k0, half);
    v16bf g0 = ldfrag(w0, k0, half), g1 = ldfrag(w1, k0, half);
    v16bf g2 = ldfrag(w2, k0, half), g3 = ldfrag(w3, k0, half);
    acc0 = wmma_bf16(a, f0, acc0);
    acc1 = wmma_bf16(a, f1, acc1);
    acc2 = wmma_bf16(a, f2, acc2);
    acc3 = wmma_bf16(a, f3, acc3);
    a = an; f0 = g0; f1 = g1; f2 = g2; f3 = g3;
  }
  acc0 = wmma_bf16(a, f0, acc0);
  acc1 = wmma_bf16(a, f1, acc1);
  acc2 = wmma_bf16(a, f2, acc2);
  acc3 = wmma_bf16(a, f3, acc3);

  #pragma unroll
  for (int t = 0; t < 4; t++) {
    v8f acc = (t == 0) ? acc0 : (t == 1) ? acc1 : (t == 2) ? acc2 : acc3;
    #pragma unroll
    for (int i = 0; i < 8; i++) {
      int row = rowBase + i + 8 * half;
      int col = colBase + t * 16 + nn;
      size_t idx = (size_t)row * DM_ + col;
      y[idx] = acc[i] + bias[col] + resid[idx];
    }
  }
}

// ---- kernel 5: LayerNorm over last dim (512), eps = 1e-3 -------------------
__global__ void __launch_bounds__(256)
layernorm_k(const float* __restrict__ y,
            const float* __restrict__ gamma,
            const float* __restrict__ beta,
            float* __restrict__ out) {
  const int row = blockIdx.x;
  const int t = threadIdx.x;
  const float* xr = y + (size_t)row * DM_;
  float x0 = xr[t], x1 = xr[t + 256];
  float s = x0 + x1, ss = x0 * x0 + x1 * x1;
  #pragma unroll
  for (int d = 1; d < 32; d <<= 1) {
    s  += __shfl_xor(s, d);
    ss += __shfl_xor(ss, d);
  }
  __shared__ float rs[8], rss[8];
  int w = t >> 5, lane = t & 31;
  if (lane == 0) { rs[w] = s; rss[w] = ss; }
  __syncthreads();
  if (w == 0) {
    float a  = (lane < 8) ? rs[lane]  : 0.f;
    float bq = (lane < 8) ? rss[lane] : 0.f;
    #pragma unroll
    for (int d = 1; d < 8; d <<= 1) { a += __shfl_xor(a, d); bq += __shfl_xor(bq, d); }
    if (lane == 0) { rs[0] = a; rss[0] = bq; }
  }
  __syncthreads();
  float mu  = rs[0]  * (1.0f / DM_);
  float var = rss[0] * (1.0f / DM_) - mu * mu;
  float inv = rsqrtf(var + 1e-3f);
  out[(size_t)row * DM_ + t]       = gamma[t]       * (x0 - mu) * inv + beta[t];
  out[(size_t)row * DM_ + t + 256] = gamma[t + 256] * (x1 - mu) * inv + beta[t + 256];
}

// ---------------------------------------------------------------------------
extern "C" void kernel_launch(void* const* d_in, const int* in_sizes, int n_in,
                              void* d_out, int out_size, void* d_ws, size_t ws_size,
                              hipStream_t stream) {
  const float* queries = (const float*)d_in[0];
  const float* keys    = (const float*)d_in[1];
  const float* values  = (const float*)d_in[2];
  const float* attW    = (const float*)d_in[3];
  const float* Wq      = (const float*)d_in[4];
  const float* bq      = (const float*)d_in[5];
  const float* Wk      = (const float*)d_in[6];
  const float* bk      = (const float*)d_in[7];
  const float* Wv      = (const float*)d_in[8];
  const float* bv      = (const float*)d_in[9];
  const float* Wo      = (const float*)d_in[10];
  const float* bo      = (const float*)d_in[11];
  const float* memK    = (const float*)d_in[12];
  const float* memV    = (const float*)d_in[13];
  const float* gamma   = (const float*)d_in[14];
  const float* beta    = (const float*)d_in[15];
  float* out = (float*)d_out;

  char* ws = (char*)d_ws;
  unsigned short* Xq  = (unsigned short*)(ws + OFF_XQ);
  unsigned short* Xk  = (unsigned short*)(ws + OFF_XK);
  unsigned short* Xv  = (unsigned short*)(ws + OFF_XV);
  unsigned short* WqT = (unsigned short*)(ws + OFF_WQT);
  unsigned short* WkT = (unsigned short*)(ws + OFF_WKT);
  unsigned short* WvT = (unsigned short*)(ws + OFF_WVT);
  unsigned short* WoT = (unsigned short*)(ws + OFF_WOT);
  unsigned short* Qh  = (unsigned short*)(ws + OFF_QH);
  unsigned short* Kh  = (unsigned short*)(ws + OFF_KH);
  unsigned short* Vt  = (unsigned short*)(ws + OFF_VT);
  unsigned short* ctx = (unsigned short*)(ws + OFF_CTX);
  float*          y   = (float*)(ws + OFF_Y);   // aliases Xq/Xk (dead by then)

  const int nElem = B_ * NQ_ * DM_;   // 8388608

  // 1) bf16 conversions (8 elem/thread) + weight transposes + memory fill
  conv_bf16_k<<<nElem / (256 * 8), 256, 0, stream>>>(queries, Xq);
  conv_bf16_k<<<nElem / (256 * 8), 256, 0, stream>>>(keys,    Xk);
  conv_bf16_k<<<nElem / (256 * 8), 256, 0, stream>>>(values,  Xv);
  transpose_w_k<<<(DM_ * DM_) / 256, 256, 0, stream>>>(Wq, WqT);
  transpose_w_k<<<(DM_ * DM_) / 256, 256, 0, stream>>>(Wk, WkT);
  transpose_w_k<<<(DM_ * DM_) / 256, 256, 0, stream>>>(Wv, WvT);
  transpose_w_k<<<(DM_ * DM_) / 256, 256, 0, stream>>>(Wo, WoT);
  fill_mem_k<<<(B_ * H_ * 64 * 64) / 256, 256, 0, stream>>>(memK, memV, Kh, Vt);

  // 2) QKV projections (WMMA): Q -> [b,h,q,d], K -> [b,h,key,d], V -> [b,h,d,key]
  dim3 ggrid(DM_ / 64, (B_ * NQ_) / 128);
  gemm_proj_k<<<ggrid, 256, 0, stream>>>(Xq, WqT, bq, Qh,
      (long)(NQ_ * DK_),  (long)DK_, 1L);
  gemm_proj_k<<<ggrid, 256, 0, stream>>>(Xk, WkT, bk, Kh,
      (long)(NKP_ * DK_), (long)DK_, 1L);
  gemm_proj_k<<<ggrid, 256, 0, stream>>>(Xv, WvT, bv, Vt,
      (long)(DK_ * NKP_), 1L, (long)NKP_);

  // 3) flash attention with memory slots + weight modulation
  attn_k<<<B_ * H_ * (NQ_ / 128), 256, 0, stream>>>(Qh, Kh, Vt, attW, ctx);

  // 4) output projection + residual (f32)
  gemm_out_k<<<ggrid, 256, 0, stream>>>(ctx, WoT, bo, queries, y);

  // 5) LayerNorm -> d_out
  layernorm_k<<<B_ * NQ_, 256, 0, stream>>>(y, gamma, beta, out);
}
